// DeformConvBlock_11647951307516
// MI455X (gfx1250) — matrix operand order repaired and checked by reading, not compiled, not run on hardware
//
#include <hip/hip_runtime.h>
#include <hip/hip_bf16.h>

typedef __attribute__((ext_vector_type(16))) _Float16 v16h;
typedef __attribute__((ext_vector_type(8)))  float    v8f;
typedef __attribute__((ext_vector_type(4)))  int      v4i;

// Address-space-qualified pointer element types for the async builtin.
typedef __attribute__((address_space(1))) v4i v4i_global;
typedef __attribute__((address_space(3))) v4i v4i_lds;

#define Bdim  8
#define CIN   128
#define COUT  128
#define Hdim  64
#define Wdim  64
#define K2    9
#define HW    (Hdim * Wdim)          // 4096
#define APAD  8
#define ASTR  (CIN + APAD)           // 136
#define BSTR  (CIN + APAD)           // 136

// ---- CDNA5 async global->LDS path (guarded; fallback = synchronous copy) ----
#if defined(__has_builtin)
#if __has_builtin(__builtin_amdgcn_global_load_async_to_lds_b128)
#define HAVE_ASYNC_B128 1
#endif
#if __has_builtin(__builtin_amdgcn_s_wait_asynccnt)
#define HAVE_WAIT_ASYNC_BUILTIN 1
#endif
#endif

__device__ __forceinline__ void wait_asynccnt0() {
#if defined(HAVE_WAIT_ASYNC_BUILTIN)
    __builtin_amdgcn_s_wait_asynccnt(0);
#else
    asm volatile("s_wait_asynccnt 0x0" ::: "memory");
#endif
}

// ---------------------------------------------------------------------------
// Kernel 0: convert w_conv f32 (COUT,CIN,3,3) -> f16 wA[kp][cout][cin]
// so each kernel-point's A matrix is one contiguous 32KB block.
// ---------------------------------------------------------------------------
__global__ __launch_bounds__(256) void wconvert_kernel(const float* __restrict__ wconv,
                                                       _Float16* __restrict__ wA) {
    int i = blockIdx.x * 256 + threadIdx.x;            // 9*128*128 = 147456
    if (i >= K2 * COUT * CIN) return;
    int cin  = i & (CIN - 1);
    int cout = (i >> 7) & (COUT - 1);
    int kp   = i >> 14;
    wA[i] = (_Float16)wconv[(size_t)cout * (CIN * K2) + cin * K2 + kp];
}

// ---------------------------------------------------------------------------
// Kernel 1: offset conv  (B,18,64,64) = conv3x3(x, w_offset), pad 1
// 18 output channels is not WMMA-shaped; plain f32 FMA.
// ---------------------------------------------------------------------------
__global__ __launch_bounds__(256) void offset_conv_kernel(const float* __restrict__ x,
                                                          const float* __restrict__ wofs,
                                                          float* __restrict__ off) {
    int idx = blockIdx.x * 256 + threadIdx.x;          // 8*18*4096 = 589824
    int w  = idx & (Wdim - 1);
    int h  = (idx >> 6) & (Hdim - 1);
    int oc = (idx >> 12) % (2 * K2);
    int b  = idx / ((2 * K2) * HW);

    const float* xb = x + (size_t)b * CIN * HW;
    const float* wb = wofs + (size_t)oc * CIN * K2;
    float acc = 0.0f;
    for (int c = 0; c < CIN; ++c) {
        const float* xc = xb + (size_t)c * HW;
        const float* wc = wb + c * K2;
#pragma unroll
        for (int ky = 0; ky < 3; ++ky) {
            int y = h - 1 + ky;
            if (y < 0 || y >= Hdim) continue;
#pragma unroll
            for (int kx = 0; kx < 3; ++kx) {
                int xx = w - 1 + kx;
                if (xx < 0 || xx >= Wdim) continue;
                acc = fmaf(xc[y * Wdim + xx], wc[ky * 3 + kx], acc);
            }
        }
    }
    off[idx] = acc;
}

// ---------------------------------------------------------------------------
// Kernel 2: deformable conv via WMMA implicit GEMM.
// One block per (b,h): 128 cout x 64 pixels. 8 waves; wave m owns cout tile m.
// Per kernel point:
//   - issue async global->LDS DMA of the 32KB A-matrix (weights for kp)
//   - overlap with the bilinear gather building B^T in LDS (f16)
//   - s_wait_asynccnt 0, barrier, then 4 ntile x 4 kchunk WMMAs per wave.
// ---------------------------------------------------------------------------
__global__ __launch_bounds__(256) void deform_wmma_kernel(const float* __restrict__ x,
                                                          const float* __restrict__ off,
                                                          const _Float16* __restrict__ wA,
                                                          float* __restrict__ out) {
    __shared__ _Float16 Amat[COUT][ASTR];   // 128 x 136 f16 = 34816 B
    __shared__ _Float16 BmatT[Wdim][BSTR];  //  64 x 136 f16 = 17408 B
    __shared__ int      sIdx[Wdim][4];      // bilinear neighbor flat indices
    __shared__ float    sW[Wdim][4];        // validity-folded bilinear weights

    const int tid  = threadIdx.x;
    const int b    = blockIdx.x / Hdim;
    const int h    = blockIdx.x % Hdim;
    const int wave = tid >> 5;
    const int lane = tid & 31;
    const int hh   = lane >> 4;     // lane half (0/1)
    const int l    = lane & 15;

    const float* xb = x + (size_t)b * CIN * HW;

    v8f acc[4];
#pragma unroll
    for (int nt = 0; nt < 4; ++nt) {
        v8f z = {0.f, 0.f, 0.f, 0.f, 0.f, 0.f, 0.f, 0.f};
        acc[nt] = z;
    }

    for (int kp = 0; kp < K2; ++kp) {
        const int ky = kp / 3, kx = kp % 3;

        // ---- sampling parameters: one thread per output pixel ----
        if (tid < Wdim) {
            int n = tid;
            float dy = off[(((size_t)b * (2 * K2) + 2 * kp) * Hdim + h) * Wdim + n];
            float dx = off[(((size_t)b * (2 * K2) + 2 * kp + 1) * Hdim + h) * Wdim + n];
            float py = (float)(h - 1 + ky) + dy;
            float px = (float)(n - 1 + kx) + dx;
            float y0f = floorf(py), x0f = floorf(px);
            int y0 = (int)y0f, x0 = (int)x0f;
            float wy1 = py - y0f, wx1 = px - x0f;
            float wy0 = 1.0f - wy1, wx0 = 1.0f - wx1;
            int y1 = y0 + 1, x1 = x0 + 1;
            float vy0 = (y0 >= 0 && y0 < Hdim) ? 1.f : 0.f;
            float vy1 = (y1 >= 0 && y1 < Hdim) ? 1.f : 0.f;
            float vx0 = (x0 >= 0 && x0 < Wdim) ? 1.f : 0.f;
            float vx1 = (x1 >= 0 && x1 < Wdim) ? 1.f : 0.f;
            int y0c = min(max(y0, 0), Hdim - 1), y1c = min(max(y1, 0), Hdim - 1);
            int x0c = min(max(x0, 0), Wdim - 1), x1c = min(max(x1, 0), Wdim - 1);
            sIdx[n][0] = y0c * Wdim + x0c;
            sIdx[n][1] = y0c * Wdim + x1c;
            sIdx[n][2] = y1c * Wdim + x0c;
            sIdx[n][3] = y1c * Wdim + x1c;
            sW[n][0] = wy0 * wx0 * vy0 * vx0;
            sW[n][1] = wy0 * wx1 * vy0 * vx1;
            sW[n][2] = wy1 * wx0 * vy1 * vx0;
            sW[n][3] = wy1 * wx1 * vy1 * vx1;
        }
        __syncthreads();

        // ---- A matrix: contiguous 32KB f16 block -> LDS (128B/thread) ----
        // Async DMA (ASYNCcnt) so it overlaps the bilinear gather below.
        {
            const _Float16* src = wA + (size_t)kp * COUT * CIN;
#if defined(HAVE_ASYNC_B128)
#pragma unroll
            for (int i = 0; i < 8; ++i) {
                int e   = tid * 64 + i * 8;             // 8 halfs (16B) per op
                int row = e >> 7, col = e & (CIN - 1);
                __builtin_amdgcn_global_load_async_to_lds_b128(
                    (v4i_global*)(uintptr_t)(src + e),
                    (v4i_lds*)(uintptr_t)&Amat[row][col],
                    0, 0);
            }
#else
#pragma unroll
            for (int i = 0; i < 8; ++i) {
                int e   = tid * 64 + i * 8;
                int row = e >> 7, col = e & (CIN - 1);
                const uint4 v = *(const uint4*)(src + e);
                *(uint4*)&Amat[row][col] = v;
            }
#endif
        }

        // ---- B matrix: bilinear gather, BmatT[pixel][cin] in f16 ----
        {
            const int n     = tid & (Wdim - 1);
            const int cbase = (tid >> 6) * 32;
            const int i00 = sIdx[n][0], i01 = sIdx[n][1];
            const int i10 = sIdx[n][2], i11 = sIdx[n][3];
            const float w00 = sW[n][0], w01 = sW[n][1];
            const float w10 = sW[n][2], w11 = sW[n][3];
            const float* xc = xb + (size_t)cbase * HW;
#pragma unroll 4
            for (int c = 0; c < 32; ++c) {
                float s = w00 * xc[i00] + w01 * xc[i01] +
                          w10 * xc[i10] + w11 * xc[i11];
                BmatT[n][cbase + c] = (_Float16)s;
                xc += HW;
            }
        }
#if defined(HAVE_ASYNC_B128)
        wait_asynccnt0();   // A-matrix DMA complete before the barrier
#endif
        __syncthreads();

        // ---- WMMA: wave 'wave' computes cout rows [wave*16, wave*16+16) ----
        {
            const _Float16* Ar = &Amat[wave * 16 + l][0];

            // A fragments for the 4 K-chunks (reused across ntile)
            v16h afr[4];
#pragma unroll
            for (int kk = 0; kk < 4; ++kk) {
                const int kb = kk * 32;
#pragma unroll
                for (int v = 0; v < 8; ++v) {           // K hh*16 .. hh*16+15 of chunk
                    int K0 = kb + hh * 16 + 2 * v;
                    afr[kk][2 * v]     = Ar[K0];
                    afr[kk][2 * v + 1] = Ar[K0 + 1];
                }
            }

#pragma unroll
            for (int nt = 0; nt < 4; ++nt) {
                const _Float16* Br = &BmatT[nt * 16 + l][0];
#pragma unroll
                for (int kk = 0; kk < 4; ++kk) {
                    const int kb = kk * 32;
                    v16h bfr;
#pragma unroll
                    for (int v = 0; v < 8; ++v) {
                        int K0 = kb + hh * 16 + 2 * v;
                        bfr[2 * v]     = Br[K0];
                        bfr[2 * v + 1] = Br[K0 + 1];
                    }
                    acc[nt] = __builtin_amdgcn_wmma_f32_16x16x32_f16(
                        false, afr[kk], false, bfr, (short)0, acc[nt],
                        false, false);
                }
            }
        }
        __syncthreads();   // protect LDS before next kp overwrites it
    }

    // ---- store: C/D layout -> out[b][cout][h][w] ----
    float* outp = out + ((size_t)b * COUT) * HW + h * Wdim;
#pragma unroll
    for (int nt = 0; nt < 4; ++nt) {
#pragma unroll
        for (int r = 0; r < 8; ++r) {
            int m = wave * 16 + r + 8 * hh;
            outp[(size_t)m * HW + nt * 16 + l] = acc[nt][r];
        }
    }
}

// ---------------------------------------------------------------------------
extern "C" void kernel_launch(void* const* d_in, const int* in_sizes, int n_in,
                              void* d_out, int out_size, void* d_ws, size_t ws_size,
                              hipStream_t stream) {
    const float* x       = (const float*)d_in[0];
    const float* w_off   = (const float*)d_in[1];
    const float* w_conv  = (const float*)d_in[2];
    float*       out     = (float*)d_out;

    // workspace: offsets (B*18*H*W f32 = 2359296 B) then wA f16 (294912 B)
    float*    off = (float*)d_ws;
    _Float16* wA  = (_Float16*)((char*)d_ws + (size_t)Bdim * 2 * K2 * HW * sizeof(float));

    wconvert_kernel<<<(K2 * COUT * CIN + 255) / 256, 256, 0, stream>>>(w_conv, wA);
    offset_conv_kernel<<<(Bdim * 2 * K2 * HW) / 256, 256, 0, stream>>>(x, w_off, off);
    deform_wmma_kernel<<<Bdim * Hdim, 256, 0, stream>>>(x, off, wA, out);
}
